// AFeFETLinear_74302934221188
// MI455X (gfx1250) — matrix operand order repaired and checked by reading, not compiled, not run on hardware
//
#include <hip/hip_runtime.h>
#include <math.h>

typedef __attribute__((ext_vector_type(16))) _Float16 v16h;
typedef __attribute__((ext_vector_type(8)))  _Float16 v8h;
typedef __attribute__((ext_vector_type(4)))  _Float16 v4h;
typedef __attribute__((ext_vector_type(8)))  float    v8f;

#define BM 128
#define BN 256          // async kernel block-N (wave tile 64x64, 8 waves 2x4)
#define BNF 128         // fallback kernel block-N
#define BK 32
#define LDST 40         // LDS row stride in f16 elements (BK + 8 pad -> 80B rows)

// --- CDNA5 async global->LDS copy (ASYNCcnt-tracked), via inline asm --------
__device__ __forceinline__ unsigned lds_addr32(const void* p) {
    // generic pointer to LDS = {SHARED_BASE[63:32], lds_byte_addr[31:0]}
    return (unsigned)(size_t)p;
}
__device__ __forceinline__ void async_ld_b128(unsigned lds, unsigned long long gaddr) {
    asm volatile("global_load_async_to_lds_b128 %0, %1, off"
                 :: "v"(lds), "v"(gaddr) : "memory");
}
__device__ __forceinline__ void wait_async0() {
    asm volatile("s_wait_asynccnt 0x0" ::: "memory");
}

// ---------------------------------------------------------------------------
// Kernel 1: FeFET weight quantization -> f16 qw in workspace
// ---------------------------------------------------------------------------
__global__ __launch_bounds__(256)
void fefet_quant_kernel(const float* __restrict__ w,
                        const int*   __restrict__ wc,
                        const float* __restrict__ alpha,
                        _Float16*    __restrict__ qw,
                        int n4)
{
    int i = blockIdx.x * blockDim.x + threadIdx.x;
    if (i >= n4) return;

    const float a    = alpha[0];
    const float inva = 1.0f / a;

    const float4 wv = reinterpret_cast<const float4*>(w)[i];
    const int4   cv = reinterpret_cast<const int4*>(wc)[i];
    float wl[4] = {wv.x, wv.y, wv.z, wv.w};
    int   cl[4] = {cv.x, cv.y, cv.z, cv.w};

    v4h o;
#pragma unroll
    for (int j = 0; j < 4; ++j) {
        float t   = 4.5f * fmaf(wl[j], a, 1.0f);
        float idx = fminf(fmaxf(rintf((t - 3.5f) * 2.0f), 0.0f), 4.0f);
        float qv  = fmaf(idx, 0.5f, 3.5f);
        float q   = (qv * (1.0f / 4.5f) - 1.0f) * inva;
        q *= 0.999000499833375f;                       // exp(-1/1000)
        float deg = fminf(fmaxf((float)cl[j] * 1e-9f, 0.0f), 0.5f);
        o[j] = (_Float16)(q * (1.0f - deg));
    }
    reinterpret_cast<v4h*>(qw)[i] = o;
}

// ---------------------------------------------------------------------------
// Kernel 2: x f32 -> f16 (one pass; halves A-side HBM traffic for the GEMM)
// ---------------------------------------------------------------------------
__global__ __launch_bounds__(256)
void fefet_xcvt_kernel(const float* __restrict__ x, _Float16* __restrict__ xh, int n4)
{
    int i = blockIdx.x * blockDim.x + threadIdx.x;
    if (i >= n4) return;
    float4 v = reinterpret_cast<const float4*>(x)[i];
    v4h o;
    o[0] = (_Float16)v.x; o[1] = (_Float16)v.y;
    o[2] = (_Float16)v.z; o[3] = (_Float16)v.w;
    reinterpret_cast<v4h*>(xh)[i] = o;
}

// ---------------------------------------------------------------------------
// Kernel 3 (primary): async double-buffered WMMA GEMM, A and B both f16
//   y[M,N] = xh[M,K] * qw[N,K]^T
//   128x256 block tile, BK=32, 8 waves (2x4), wave tile 64x64
//   -> 16 WMMAs and 16 ds_load_b128 per barrier (ds:wmma = 1:1)
// ---------------------------------------------------------------------------
__global__ __launch_bounds__(256)
void fefet_gemm_async_kernel(const _Float16* __restrict__ xh,
                             const _Float16* __restrict__ qw,
                             float*          __restrict__ y,
                             int M, int N, int K)
{
    __shared__ __align__(16) _Float16 As[2][BM * LDST];   // 2 x 10 KB
    __shared__ __align__(16) _Float16 Bs[2][BN * LDST];   // 2 x 20 KB

    const int tid  = threadIdx.x;
    const int lane = tid & 31;
    const int wave = tid >> 5;

    const int m0 = blockIdx.y * BM;
    const int n0 = blockIdx.x * BN;
    const int wm0 = (wave >> 2) * 64;   // 2 waves in M
    const int wn0 = (wave & 3) * 64;    // 4 waves in N

    // A staging: 2 threads per row, 32B (2 x b128) each
    const int afrow = tid >> 1;
    const int afhal = tid & 1;
    // B staging: 1 thread per row, 64B (4 x b128)
    const int bfrow = tid;

    const unsigned long long ga =
        (unsigned long long)(size_t)(xh + (size_t)(m0 + afrow) * K + afhal * 16);
    const unsigned long long gb =
        (unsigned long long)(size_t)(qw + (size_t)(n0 + bfrow) * K);
    const unsigned la[2] = { lds_addr32(&As[0][afrow * LDST + afhal * 16]),
                             lds_addr32(&As[1][afrow * LDST + afhal * 16]) };
    const unsigned lb[2] = { lds_addr32(&Bs[0][bfrow * LDST]),
                             lds_addr32(&Bs[1][bfrow * LDST]) };

    const int lrow  = lane & 15;
    const int lhalf = lane >> 4;

    const v8f vzero = {0.f, 0.f, 0.f, 0.f, 0.f, 0.f, 0.f, 0.f};
    v8f acc[4][4];
#pragma unroll
    for (int im = 0; im < 4; ++im)
#pragma unroll
        for (int in = 0; in < 4; ++in)
            acc[im][in] = vzero;

    auto issue_tile = [&](int buf, int k0) {
        const unsigned long long gka = ga + (unsigned long long)k0 * 2;
        const unsigned long long gkb = gb + (unsigned long long)k0 * 2;
        async_ld_b128(la[buf],      gka);
        async_ld_b128(la[buf] + 16, gka + 16);
        async_ld_b128(lb[buf],      gkb);
        async_ld_b128(lb[buf] + 16, gkb + 16);
        async_ld_b128(lb[buf] + 32, gkb + 32);
        async_ld_b128(lb[buf] + 48, gkb + 48);
    };

    const int nk = K / BK;
    issue_tile(0, 0);

    for (int kt = 0; kt < nk; ++kt) {
        const int buf = kt & 1;

        wait_async0();        // this wave's async LDS writes have landed
        __syncthreads();      // everyone's writes landed; prev-iter reads drained

        if (kt + 1 < nk) issue_tile(buf ^ 1, (kt + 1) * BK);
        if (kt + 2 < nk) {    // warm GL2 two tiles ahead
            __builtin_prefetch((const void*)(size_t)(ga + (size_t)(kt + 2) * BK * 2), 0, 1);
            __builtin_prefetch((const void*)(size_t)(gb + (size_t)(kt + 2) * BK * 2), 0, 1);
        }

        // ---- A fragments: 16-bit 16x32 layout (ISA 7.12.2) ----
        // lane<16: row=lane, K = {0..7,16..23}; lane>=16: same row, K = {8..15,24..31}
        v16h afr[4];
#pragma unroll
        for (int im = 0; im < 4; ++im) {
            const _Float16* ar = &As[buf][(wm0 + im * 16 + lrow) * LDST];
            v8h c0 = *(const v8h*)(ar + lhalf * 8);
            v8h c1 = *(const v8h*)(ar + 16 + lhalf * 8);
            afr[im] = __builtin_shufflevector(c0, c1,
                0,1,2,3,4,5,6,7,8,9,10,11,12,13,14,15);
        }
        // ---- B fragments: 32x16, lane = N column, K = {0..7,16..23} / {8..15,24..31} by lane>>4 ----
        v16h bfr[4];
#pragma unroll
        for (int in = 0; in < 4; ++in) {
            const _Float16* br = &Bs[buf][(wn0 + in * 16 + lrow) * LDST + lhalf * 8];
            v8h c0 = *(const v8h*)(br + 0);
            v8h c1 = *(const v8h*)(br + 16);
            bfr[in] = __builtin_shufflevector(c0, c1,
                0,1,2,3,4,5,6,7,8,9,10,11,12,13,14,15);
        }

#pragma unroll
        for (int im = 0; im < 4; ++im)
#pragma unroll
            for (int in = 0; in < 4; ++in)
                acc[im][in] = __builtin_amdgcn_wmma_f32_16x16x32_f16(
                    false, afr[im], false, bfr[in],
                    (short)0, acc[im][in], false, false);
    }

    // ---- epilogue: C/D layout -> row = vgpr + 8*(lane>=16), col = lane&15 ----
    const int orow = lhalf * 8;
    const int ocol = lrow;
#pragma unroll
    for (int im = 0; im < 4; ++im) {
#pragma unroll
        for (int in = 0; in < 4; ++in) {
            float* dst = y + (size_t)(m0 + wm0 + im * 16 + orow) * N
                           + (n0 + wn0 + in * 16 + ocol);
            v8f a = acc[im][in];
#pragma unroll
            for (int r = 0; r < 8; ++r)
                dst[(size_t)r * N] = a[r];
        }
    }
}

// ---------------------------------------------------------------------------
// Kernel 3 (fallback, ws too small for f16 x): register-staged GEMM, A = f32
// ---------------------------------------------------------------------------
__global__ __launch_bounds__(256)
void fefet_gemm_kernel(const float*    __restrict__ x,
                       const _Float16* __restrict__ qw,
                       float*          __restrict__ y,
                       int M, int N, int K)
{
    __shared__ __align__(16) _Float16 As[BM * LDST];
    __shared__ __align__(16) _Float16 Bs[BNF * LDST];

    const int tid  = threadIdx.x;
    const int lane = tid & 31;
    const int wave = tid >> 5;
    const int m0 = blockIdx.y * BM;
    const int n0 = blockIdx.x * BNF;
    const int wm0 = (wave >> 2) * 64;
    const int wn0 = (wave & 3) * 32;
    const int frow = tid >> 1;
    const int fhal = tid & 1;

    const float*    xg = x  + (size_t)(m0 + frow) * K + fhal * 16;
    const _Float16* bg = qw + (size_t)(n0 + frow) * K + fhal * 16;
    _Float16* asto = &As[frow * LDST + fhal * 16];
    _Float16* bsto = &Bs[frow * LDST + fhal * 16];

    const int lrow  = lane & 15;
    const int lhalf = lane >> 4;

    const v8f vzero = {0.f, 0.f, 0.f, 0.f, 0.f, 0.f, 0.f, 0.f};
    v8f acc[4][2];
#pragma unroll
    for (int im = 0; im < 4; ++im)
#pragma unroll
        for (int in = 0; in < 2; ++in)
            acc[im][in] = vzero;

    for (int k0 = 0; k0 < K; k0 += BK) {
        float4 a0 = *(const float4*)(xg + k0 + 0);
        float4 a1 = *(const float4*)(xg + k0 + 4);
        float4 a2 = *(const float4*)(xg + k0 + 8);
        float4 a3 = *(const float4*)(xg + k0 + 12);
        v8h    b0 = *(const v8h*)(bg + k0 + 0);
        v8h    b1 = *(const v8h*)(bg + k0 + 8);

        __syncthreads();

        v8h ha, hb;
        ha[0]=(_Float16)a0.x; ha[1]=(_Float16)a0.y; ha[2]=(_Float16)a0.z; ha[3]=(_Float16)a0.w;
        ha[4]=(_Float16)a1.x; ha[5]=(_Float16)a1.y; ha[6]=(_Float16)a1.z; ha[7]=(_Float16)a1.w;
        hb[0]=(_Float16)a2.x; hb[1]=(_Float16)a2.y; hb[2]=(_Float16)a2.z; hb[3]=(_Float16)a2.w;
        hb[4]=(_Float16)a3.x; hb[5]=(_Float16)a3.y; hb[6]=(_Float16)a3.z; hb[7]=(_Float16)a3.w;

        *(v8h*)(asto + 0) = ha;
        *(v8h*)(asto + 8) = hb;
        *(v8h*)(bsto + 0) = b0;
        *(v8h*)(bsto + 8) = b1;

        __syncthreads();

        v16h afr[4];
#pragma unroll
        for (int im = 0; im < 4; ++im) {
            const _Float16* ar = &As[(wm0 + im * 16 + lrow) * LDST];
            v8h c0 = *(const v8h*)(ar + lhalf * 8);
            v8h c1 = *(const v8h*)(ar + 16 + lhalf * 8);
            afr[im] = __builtin_shufflevector(c0, c1,
                0,1,2,3,4,5,6,7,8,9,10,11,12,13,14,15);
        }
        v16h bfr[2];
#pragma unroll
        for (int in = 0; in < 2; ++in) {
            const _Float16* br = &Bs[(wn0 + in * 16 + lrow) * LDST + lhalf * 8];
            v8h c0 = *(const v8h*)(br + 0);
            v8h c1 = *(const v8h*)(br + 16);
            bfr[in] = __builtin_shufflevector(c0, c1,
                0,1,2,3,4,5,6,7,8,9,10,11,12,13,14,15);
        }
#pragma unroll
        for (int im = 0; im < 4; ++im)
#pragma unroll
            for (int in = 0; in < 2; ++in)
                acc[im][in] = __builtin_amdgcn_wmma_f32_16x16x32_f16(
                    false, afr[im], false, bfr[in],
                    (short)0, acc[im][in], false, false);
    }

    const int orow = lhalf * 8;
    const int ocol = lrow;
#pragma unroll
    for (int im = 0; im < 4; ++im) {
#pragma unroll
        for (int in = 0; in < 2; ++in) {
            float* dst = y + (size_t)(m0 + wm0 + im * 16 + orow) * N
                           + (n0 + wn0 + in * 16 + ocol);
            v8f a = acc[im][in];
#pragma unroll
            for (int r = 0; r < 8; ++r)
                dst[(size_t)r * N] = a[r];
        }
    }
}

// ---------------------------------------------------------------------------
extern "C" void kernel_launch(void* const* d_in, const int* in_sizes, int n_in,
                              void* d_out, int out_size, void* d_ws, size_t ws_size,
                              hipStream_t stream)
{
    const float* x     = (const float*)d_in[0];   // [8,2048,4096] f32
    const float* w     = (const float*)d_in[1];   // [4096,4096]  f32
    const float* alpha = (const float*)d_in[2];   // [1] f32
    const int*   wc    = (const int*)d_in[3];     // [4096,4096] int
    float*       out   = (float*)d_out;           // [8,2048,4096] f32

    const int NK = in_sizes[1];
    const int K  = (int)llround(sqrt((double)NK));
    const int N  = K;
    const int M  = in_sizes[0] / K;

    _Float16* qw = (_Float16*)d_ws;
    const size_t qwBytes = (size_t)NK * 2;
    const size_t xhBytes = (size_t)M * (size_t)K * 2;

    // 1) quantize weights
    const int n4w = NK / 4;
    fefet_quant_kernel<<<(n4w + 255) / 256, 256, 0, stream>>>(w, wc, alpha, qw, n4w);

    if (ws_size >= qwBytes + xhBytes && (N % BN) == 0) {
        // 2) x -> f16 once, 3) async double-buffered WMMA GEMM
        _Float16* xxh = (_Float16*)((char*)d_ws + qwBytes);
        const int n4x = (int)((size_t)M * K / 4);
        fefet_xcvt_kernel<<<(n4x + 255) / 256, 256, 0, stream>>>(x, xxh, n4x);
        dim3 grid(N / BN, M / BM);
        fefet_gemm_async_kernel<<<grid, 256, 0, stream>>>(xxh, qw, out, M, N, K);
    } else {
        dim3 grid(N / BNF, M / BM);
        fefet_gemm_kernel<<<grid, 256, 0, stream>>>(x, qw, out, M, N, K);
    }
}
